// GQA_52475910422559
// MI455X (gfx1250) — matrix operand order repaired and checked by reading, not compiled, not run on hardware
//
#include <hip/hip_runtime.h>
#include <hip/hip_bf16.h>

typedef __bf16 bf16;
typedef __attribute__((ext_vector_type(16))) __bf16 v16bf;
typedef __attribute__((ext_vector_type(8)))  __bf16 v8bf;
typedef __attribute__((ext_vector_type(8)))  float  v8f;
typedef __attribute__((ext_vector_type(4)))  float  v4f;

#define LANE ((int)(threadIdx.x & 31))

__device__ inline v8f wmma_bf16(v16bf a, v16bf b, v8f c) {
  return __builtin_amdgcn_wmma_f32_16x16x32_bf16(false, a, false, b, (short)0, c,
                                                 false, false);
}

// A-fragment (16x32): A[m][k] = M[(m0+m)*ld + k0+k], f32 source -> bf16
__device__ inline v16bf load_a_f32(const float* M, int ld, int m0, int k0) {
  int lane = LANE, m = lane & 15, half = lane >> 4;
  const float* r = M + (size_t)(m0 + m) * ld + k0;
  v16bf a;
#pragma unroll
  for (int i = 0; i < 8; ++i) a[i] = (bf16)r[half * 16 + i];
#pragma unroll
  for (int i = 0; i < 8; ++i) a[8 + i] = (bf16)r[half * 16 + 8 + i];
  return a;
}

// B-fragment (32x16): B[k][n] = M[(n0+n)*ld + k0+k], f32 source -> bf16
__device__ inline v16bf load_bT_f32(const float* M, int ld, int n0, int k0) {
  int lane = LANE;
  const float* r = M + (size_t)(n0 + (lane & 15)) * ld + k0 + (lane >> 4) * 16;
  v16bf b;
#pragma unroll
  for (int i = 0; i < 16; ++i) b[i] = (bf16)r[i];
  return b;
}

// A-fragment from bf16 memory (row-major m x k)
__device__ inline v16bf load_a_bf16(const bf16* M, int ld, int m0, int k0) {
  int lane = LANE, m = lane & 15, half = lane >> 4;
  const bf16* r = M + (size_t)(m0 + m) * ld + k0;
  v8bf lo = *(const v8bf*)(r + half * 16);
  v8bf hi = *(const v8bf*)(r + half * 16 + 8);
  v16bf a;
#pragma unroll
  for (int i = 0; i < 8; ++i) { a[i] = lo[i]; a[8 + i] = hi[i]; }
  return a;
}

// B-fragment from bf16 memory: B[k][n] = M[(n0+n)*ld + k0+k]
__device__ inline v16bf load_bT_bf16(const bf16* M, int ld, int n0, int k0) {
  int lane = LANE;
  const bf16* r = M + (size_t)(n0 + (lane & 15)) * ld + k0 + (lane >> 4) * 16;
  return *(const v16bf*)r;
}

// ---------------------------------------------------------------- utilities
__global__ void cvt_f32_bf16(const float* __restrict__ src, bf16* __restrict__ dst,
                             int n) {
  int i = blockIdx.x * blockDim.x + threadIdx.x;
  if (i < n) dst[i] = (bf16)src[i];
}

// ---------------------------------------------- per-head projection, row-major out
// out[t][h*64 + d] = sum_e X[t][h*IN+e] * W[d][e]   (A = W rows, B = X^T)
__global__ void proj_rm(const float* __restrict__ X, const float* __restrict__ W,
                        bf16* __restrict__ out, int H, int IN) {
  int h = blockIdx.x % H;
  int t0 = (blockIdx.x / H) * 16;
  int lane = LANE, half = lane >> 4;
  int ldo = H * 64;
#pragma unroll
  for (int d0 = 0; d0 < 64; d0 += 16) {
    v8f c = {};
    for (int k0 = 0; k0 < IN; k0 += 32) {
      v16bf a = load_a_f32(W, IN, d0, k0);
      v16bf b = load_bT_f32(X, 512, t0, h * IN + k0);
      c = wmma_bf16(a, b, c);
    }
    v8bf s;
#pragma unroll
    for (int r = 0; r < 8; ++r) s[r] = (bf16)c[r];
    *(v8bf*)(out + (size_t)(t0 + (lane & 15)) * ldo + h * 64 + d0 + 8 * half) = s;
  }
}

// ----------------------------------- V projection, transposed out: vT[(n*4+h)*64+d][l]
__global__ void proj_v(const float* __restrict__ X, const float* __restrict__ W,
                       bf16* __restrict__ vT) {
  int h = blockIdx.x & 3;
  int t0 = (blockIdx.x >> 2) * 16;
  int nb = t0 >> 11;          // batch index (t0 tiles never straddle 2048)
  int l0 = t0 & 2047;
  int lane = LANE, half = lane >> 4;
#pragma unroll
  for (int d0 = 0; d0 < 64; d0 += 16) {
    v8f c = {};
#pragma unroll
    for (int k0 = 0; k0 < 128; k0 += 32) {
      v16bf a = load_a_f32(X, 512, t0, h * 128 + k0);   // A = tokens
      v16bf b = load_bT_f32(W, 128, d0, k0);            // B[e][d] = W[d][e]
      c = wmma_bf16(a, b, c);
    }
    v8bf s;
#pragma unroll
    for (int r = 0; r < 8; ++r) s[r] = (bf16)c[r];
    size_t row = (size_t)((nb * 4 + h) * 64 + d0 + (lane & 15));
    *(v8bf*)(vT + row * 2048 + l0 + 8 * half) = s;      // contiguous along tokens
  }
}

// ------------------------------------------- pass 1: row max & sum(exp) via WMMA
__global__ void attn_stats(const bf16* __restrict__ qb, const bf16* __restrict__ kb,
                           float* __restrict__ rowmax, float* __restrict__ rowsum) {
  int q0 = (blockIdx.x & 127) * 16;
  int head = blockIdx.x >> 7;                 // n*8 + qh
  int n = head >> 3, qh = head & 7, kvh = qh & 3;
  int lane = LANE, half = lane >> 4;
  const bf16* Q = qb + (size_t)n * 2048 * 512 + qh * 64;
  const bf16* K = kb + (size_t)n * 2048 * 256 + kvh * 64;
  v16bf qa0 = load_a_bf16(Q, 512, q0, 0);
  v16bf qa1 = load_a_bf16(Q, 512, q0, 32);
  const float scale = 0.04419417382415922f;   // 1/sqrt(512)
  float mrun[8], srun[8];
#pragma unroll
  for (int r = 0; r < 8; ++r) { mrun[r] = -3.0e38f; srun[r] = 0.f; }
  for (int kt = 0; kt < 2048; kt += 16) {
    v16bf b0 = load_bT_bf16(K, 256, kt, 0);
    v16bf b1 = load_bT_bf16(K, 256, kt, 32);
    v8f s = {};
    s = wmma_bf16(qa0, b0, s);
    s = wmma_bf16(qa1, b1, s);
#pragma unroll
    for (int r = 0; r < 8; ++r) {
      float x = s[r] * scale;
      float nm = fmaxf(mrun[r], x);
      srun[r] = srun[r] * __expf(mrun[r] - nm) + __expf(x - nm);
      mrun[r] = nm;
    }
  }
  // combine the 16 lanes of each half (each row lives in one 16-lane half)
#pragma unroll
  for (int off = 1; off < 16; off <<= 1) {
#pragma unroll
    for (int r = 0; r < 8; ++r) {
      float om = __shfl_xor(mrun[r], off, 32);
      float os = __shfl_xor(srun[r], off, 32);
      float nm = fmaxf(mrun[r], om);
      srun[r] = srun[r] * __expf(mrun[r] - nm) + os * __expf(om - nm);
      mrun[r] = nm;
    }
  }
  if ((lane & 15) == 0) {
    size_t base = (size_t)head * 2048 + q0 + 8 * half;
#pragma unroll
    for (int r = 0; r < 8; ++r) { rowmax[base + r] = mrun[r]; rowsum[base + r] = srun[r]; }
  }
}

// -------------- pass 2: normalized attn (f32, stored once, coalesced) + O = attn @ V
__global__ void attn_out(const bf16* __restrict__ qb, const bf16* __restrict__ kb,
                         const bf16* __restrict__ vT,
                         const float* __restrict__ rowmax,
                         const float* __restrict__ rowsum,
                         float* __restrict__ attn, bf16* __restrict__ opre) {
  __shared__ __attribute__((aligned(16))) float pf[16 * 32];  // f32 P tile (2 KB)
  __shared__ __attribute__((aligned(16))) bf16  ot[16 * 64];  // O epilogue tile (2 KB)
  int q0 = (blockIdx.x & 127) * 16;
  int head = blockIdx.x >> 7;
  int n = head >> 3, qh = head & 7, kvh = qh & 3;
  int lane = LANE, half = lane >> 4, col = lane & 15;
  const bf16* Q = qb + (size_t)n * 2048 * 512 + qh * 64;
  const bf16* K = kb + (size_t)n * 2048 * 256 + kvh * 64;
  const bf16* V = vT + (size_t)(n * 4 + kvh) * 64 * 2048;
  v16bf qa0 = load_a_bf16(Q, 512, q0, 0);
  v16bf qa1 = load_a_bf16(Q, 512, q0, 32);
  const float scale = 0.04419417382415922f;
  float mx[8], inv[8];
  {
    size_t base = (size_t)head * 2048 + q0 + 8 * half;
#pragma unroll
    for (int r = 0; r < 8; ++r) { mx[r] = rowmax[base + r]; inv[r] = 1.0f / rowsum[base + r]; }
  }
  v8f o0 = {}, o1 = {}, o2 = {}, o3 = {};
  float* ab = attn + (size_t)head * 2048 * 2048 + (size_t)q0 * 2048;
  for (int k0 = 0; k0 < 2048; k0 += 32) {
    if (k0 + 32 < 2048)  // gfx1250 global_prefetch_b8 on next K tile
      __builtin_prefetch(K + (size_t)(k0 + 32 + col) * 256, 0, 0);
#pragma unroll
    for (int sub = 0; sub < 2; ++sub) {
      int kt = k0 + sub * 16;
      v16bf b0 = load_bT_bf16(K, 256, kt, 0);
      v16bf b1 = load_bT_bf16(K, 256, kt, 32);
      v8f s = {};
      s = wmma_bf16(qa0, b0, s);
      s = wmma_bf16(qa1, b1, s);
#pragma unroll
      for (int r = 0; r < 8; ++r) {
        float p = __expf(s[r] * scale - mx[r]) * inv[r];
        pf[(8 * half + r) * 32 + sub * 16 + col] = p;  // stage normalized P in LDS
      }
    }
    // Coalesced attn store: 16x32 f32 tile, 8 lanes cover a 128B row segment.
    // Single-wave block: LDS ops are in-order, no barrier needed.
#pragma unroll
    for (int pass = 0; pass < 4; ++pass) {
      int idx = pass * 128 + lane * 4;       // element in 16x32 tile
      int rr = idx >> 5, cc = idx & 31;
      v4f t = *(const v4f*)(pf + rr * 32 + cc);
      *(v4f*)(ab + (size_t)rr * 2048 + k0 + cc) = t;
    }
    // Reload P as 16x32 A-fragment (f32 LDS -> bf16) and accumulate O = P @ V.
    v16bf pa  = load_a_f32(pf, 32, 0, 0);
    v16bf vb0 = load_bT_bf16(V, 2048, 0,  k0);
    v16bf vb1 = load_bT_bf16(V, 2048, 16, k0);
    v16bf vb2 = load_bT_bf16(V, 2048, 32, k0);
    v16bf vb3 = load_bT_bf16(V, 2048, 48, k0);
    o0 = wmma_bf16(pa, vb0, o0);
    o1 = wmma_bf16(pa, vb1, o1);
    o2 = wmma_bf16(pa, vb2, o2);
    o3 = wmma_bf16(pa, vb3, o3);
  }
  // Epilogue: stage 16x64 bf16 O tile in LDS, then coalesced 16B stores.
#pragma unroll
  for (int r = 0; r < 8; ++r) {
    int m = 8 * half + r;
    ot[m * 64 + col +  0] = (bf16)o0[r];
    ot[m * 64 + col + 16] = (bf16)o1[r];
    ot[m * 64 + col + 32] = (bf16)o2[r];
    ot[m * 64 + col + 48] = (bf16)o3[r];
  }
  int t0 = n * 2048 + q0;
#pragma unroll
  for (int pass = 0; pass < 4; ++pass) {
    int idx = pass * 256 + lane * 8;          // element in 16x64 tile
    int rr = idx >> 6, cc = idx & 63;
    v8bf t = *(const v8bf*)(ot + rr * 64 + cc);
    *(v8bf*)(opre + (size_t)(t0 + rr) * 512 + qh * 64 + cc) = t;
  }
}

// ------------------------------------- out = opre @ Wo^T + bo   (A = Wo, B = opre^T)
__global__ void out_proj(const bf16* __restrict__ opre, const bf16* __restrict__ wo,
                         const float* __restrict__ bo, float* __restrict__ out) {
  int j0 = (blockIdx.x & 31) * 16;
  int t0 = (blockIdx.x >> 5) * 16;
  int lane = LANE, half = lane >> 4, col = lane & 15;
  v8f c = {};
#pragma unroll 4
  for (int k0 = 0; k0 < 512; k0 += 32) {
    v16bf a = load_a_bf16(wo, 512, j0, k0);
    v16bf b = load_bT_bf16(opre, 512, t0, k0);
    c = wmma_bf16(a, b, c);
  }
  int j = j0 + 8 * half;
  v8f res;
#pragma unroll
  for (int r = 0; r < 8; ++r) res[r] = c[r] + bo[j + r];
  *(v8f*)(out + (size_t)(t0 + col) * 512 + j) = res;   // 32B contiguous per lane
}

extern "C" void kernel_launch(void* const* d_in, const int* in_sizes, int n_in,
                              void* d_out, int out_size, void* d_ws, size_t ws_size,
                              hipStream_t stream) {
  (void)in_sizes; (void)n_in; (void)out_size; (void)ws_size;
  const float* values  = (const float*)d_in[0];
  const float* keys    = (const float*)d_in[1];
  const float* queries = (const float*)d_in[2];
  const float* Wv      = (const float*)d_in[3];
  const float* Wk      = (const float*)d_in[4];
  const float* Wq      = (const float*)d_in[5];
  const float* Wo      = (const float*)d_in[6];
  const float* bo      = (const float*)d_in[7];

  float* out  = (float*)d_out;
  float* attn = out + (size_t)2 * 2048 * 512;          // outputs concatenated: out, attn

  char* ws = (char*)d_ws;
  bf16*  qb   = (bf16*)(ws);                           // [4096][512]  4 MB
  bf16*  kb   = (bf16*)(ws + 4194304);                 // [4096][256]  2 MB
  bf16*  vT   = (bf16*)(ws + 6291456);                 // [2*4*64][2048] 2 MB
  bf16*  opre = (bf16*)(ws + 8388608);                 // [4096][512]  4 MB
  bf16*  wo   = (bf16*)(ws + 12582912);                // [512][512]   0.5 MB
  float* rmax = (float*)(ws + 13107200);               // [2*8*2048]
  float* rsum = (float*)(ws + 13238272);               // [2*8*2048]

  cvt_f32_bf16<<<1024, 256, 0, stream>>>(Wo, wo, 512 * 512);
  proj_rm   <<<2048, 32, 0, stream>>>(queries, Wq, qb, 8, 64);
  proj_rm   <<<1024, 32, 0, stream>>>(keys,    Wk, kb, 4, 128);
  proj_v    <<<1024, 32, 0, stream>>>(values,  Wv, vT);
  attn_stats<<<2048, 32, 0, stream>>>(qb, kb, rmax, rsum);
  attn_out  <<<2048, 32, 0, stream>>>(qb, kb, vT, rmax, rsum, attn, opre);
  out_proj  <<<8192, 32, 0, stream>>>(opre, wo, bo, out);
}
